// Baseline_73607149519542
// MI455X (gfx1250) — compile-verified, matrix operand order checked by reading, not run on hardware
//
#include <hip/hip_runtime.h>

// Problem constants (fixed by the reference).
#define B_SEG  256   // batch / #segments
#define C_IN   128   // input channels
#define H_DIM  256   // hidden size
#define NCLS   400   // classes
#define TSTEPS 256   // only h_outs[:B] feed the output -> 256 steps suffice

#define STEP_GEMM_BLKS 32            // 256 waves = 16 bTiles x 16 hTiles
#define STEP_RED_BLK   32            // block id doing the fused reduction
#define STEP_CVT_BLK0  33            // blocks converting x_{t+1}
#define STEP_NBLKS     41            // 32 gemm + 1 reduce + 8 convert

typedef __attribute__((ext_vector_type(16))) __bf16 v16bf;
typedef __attribute__((ext_vector_type(8)))  float  v8f;

union Frag {
  v16bf v;
  uint4 q[2];
};

// 16-bit A/B fragment per ISA 7.12.2: a lane holds two runs of 8 consecutive
// K values starting at half*8 and 16+half*8 (half = lane>>4).
__device__ __forceinline__ v16bf frag_from_bf(const __bf16* row, int half) {
  Frag f;
  f.q[0] = *(const uint4*)(row + half * 8);        // 16B-aligned runs
  f.q[1] = *(const uint4*)(row + 16 + half * 8);
  return f.v;
}

// f32 source -> bf16 fragment using native cvt instructions.
__device__ __forceinline__ v16bf frag_from_f32(const float* row, int half) {
  float4 a0 = ((const float4*)(row + half * 8))[0];
  float4 a1 = ((const float4*)(row + half * 8))[1];
  float4 b0 = ((const float4*)(row + 16 + half * 8))[0];
  float4 b1 = ((const float4*)(row + 16 + half * 8))[1];
  v16bf v;
  v[0]  = (__bf16)a0.x; v[1]  = (__bf16)a0.y; v[2]  = (__bf16)a0.z; v[3]  = (__bf16)a0.w;
  v[4]  = (__bf16)a1.x; v[5]  = (__bf16)a1.y; v[6]  = (__bf16)a1.z; v[7]  = (__bf16)a1.w;
  v[8]  = (__bf16)b0.x; v[9]  = (__bf16)b0.y; v[10] = (__bf16)b0.z; v[11] = (__bf16)b0.w;
  v[12] = (__bf16)b1.x; v[13] = (__bf16)b1.y; v[14] = (__bf16)b1.z; v[15] = (__bf16)b1.w;
  return v;
}

__device__ __forceinline__ float sigmoidf_(float x) { return 1.0f / (1.0f + expf(-x)); }

// ---------------------------------------------------------------------------
// Exclusive prefix sum of segment lengths (tiny, once per launch).
__global__ void prefix_kernel(const int* __restrict__ lens, int* __restrict__ offs, int n) {
  if (blockIdx.x == 0 && threadIdx.x == 0) {
    int s = 0;
    for (int i = 0; i < n; ++i) { offs[i] = s; s += lens[i]; }
  }
}

// Per-segment SUM over the ragged input: sums[b, c].
__global__ void segsum_kernel(const float* __restrict__ x, const int* __restrict__ offs,
                              const int* __restrict__ lens, float* __restrict__ sums) {
  int b = blockIdx.x, c = threadIdx.x;
  int off = offs[b], len = lens[b];
  float s = 0.0f;
  for (int r = 0; r < len; ++r) s += x[(size_t)(off + r) * C_IN + c];
  sums[(size_t)b * C_IN + c] = s;
}

// Generic f32 -> bf16 buffer conversion (weights once; h0 once).
__global__ void cvt_bf16_kernel(const float* __restrict__ src,
                                __bf16* __restrict__ dst, int n) {
  int i = blockIdx.x * blockDim.x + threadIdx.x;
  if (i < n) dst[i] = (__bf16)src[i];
}

// Convert the time-t slice of the ragged input into a padded bf16 [B, C] tile.
__global__ void xcvt_kernel(const float* __restrict__ x, const int* __restrict__ offs,
                            const int* __restrict__ lens, int t,
                            __bf16* __restrict__ xbf) {
  int e = blockIdx.x * blockDim.x + threadIdx.x;
  if (e < B_SEG * C_IN) {
    int b = e >> 7, c = e & (C_IN - 1);
    float v = (t < lens[b]) ? x[(size_t)(offs[b] + t) * C_IN + c] : 0.0f;
    xbf[e] = (__bf16)v;
  }
}

// Masked mean over batch rows of h_t (standalone version; used for t = 255).
__global__ void reduce_h_kernel(const float* __restrict__ h, const int* __restrict__ lens,
                                int t, float* __restrict__ fcin) {
  const int hc = threadIdx.x;
  int count = lens[t];
  if (count > B_SEG) count = B_SEG;
  float s = 0.0f;
  for (int b = 0; b < count; ++b) s += h[(size_t)b * H_DIM + hc];
  fcin[(size_t)t * H_DIM + hc] = s / (float)count;
}

// ---------------------------------------------------------------------------
// Generic D[M,N] = act(A[M,K] @ W[N,K]^T + bias). One 16x16 tile per wave.
// M,N multiples of 16; K multiple of 32. A/W converted to bf16 on the fly.
__global__ void wmma_gemm_nt(const float* __restrict__ A, int lda,
                             const float* __restrict__ W, int ldw,
                             const float* __restrict__ bias,
                             float* __restrict__ D, int ldd,
                             int Nt, int total, int K, int relu) {
  const int wave = threadIdx.x >> 5;
  const int lane = threadIdx.x & 31;
  const int idx  = blockIdx.x * (blockDim.x >> 5) + wave;
  if (idx >= total) return;                 // uniform per wave -> EXEC all-1s
  const int tileM = idx / Nt, tileN = idx % Nt;
  const int half = lane >> 4, l15 = lane & 15;

  const float* arow = A + (size_t)(tileM * 16 + l15) * lda;
  const float* wrow = W + (size_t)(tileN * 16 + l15) * ldw;

  v8f acc = {};
  for (int k0 = 0; k0 < K; k0 += 32) {
    v16bf a = frag_from_f32(arow + k0, half);
    v16bf b = frag_from_f32(wrow + k0, half);
    acc = __builtin_amdgcn_wmma_f32_16x16x32_bf16(false, a, false, b, (short)0,
                                                  acc, false, false);
  }

  const int col = tileN * 16 + l15;
  const float bv = bias ? bias[col] : 0.0f;
#pragma unroll
  for (int r = 0; r < 8; ++r) {
    int row = tileM * 16 + r + half * 8;    // C/D layout: VGPR r -> M=r / r+8
    float v = acc[r] + bv;
    if (relu) v = fmaxf(v, 0.0f);
    D[(size_t)row * ldd + col] = v;
  }
}

// ---------------------------------------------------------------------------
// Fused LSTM step at time t:
//   blocks 0..31 : gates GEMM (bf16 WMMA, f32 acc) + cell update; each wave
//                  owns one 16x16 (batch, hidden) block and its 4 gate tiles.
//   block  32    : masked batch-mean of h_{t-1} -> fc_in[t-1] (indep. of GEMM)
//   blocks 33..40: convert x_{t+1} slice to padded bf16 (indep. of recurrence)
__global__ void lstm_step_kernel(const __bf16* __restrict__ xbf,   // [B,C] t slice
                                 const __bf16* __restrict__ hbf,   // [B,H] h_{t-1}
                                 const float*  __restrict__ h_in,  // f32 h_{t-1}
                                 const float*  __restrict__ c_in,
                                 const __bf16* __restrict__ WihB,
                                 const __bf16* __restrict__ WhhB,
                                 const float* __restrict__ bih, const float* __restrict__ bhh,
                                 float* __restrict__ h_out, __bf16* __restrict__ hbf_out,
                                 float* __restrict__ c_out,
                                 const float* __restrict__ x, const int* __restrict__ offs,
                                 const int* __restrict__ lens, int t,
                                 __bf16* __restrict__ xbf_next,
                                 float* __restrict__ fcin) {
  const int bid = blockIdx.x;

  if (bid == STEP_RED_BLK) {                    // fused reduction of h_{t-1}
    const int rt = t - 1;
    if (rt >= 0) {
      const int hc = threadIdx.x;               // 256 threads
      int count = lens[rt];
      if (count > B_SEG) count = B_SEG;
      float s = 0.0f;
      for (int b = 0; b < count; ++b) s += h_in[(size_t)b * H_DIM + hc];
      fcin[(size_t)rt * H_DIM + hc] = s / (float)count;
    }
    return;
  }
  if (bid >= STEP_CVT_BLK0) {                   // prepare x_{t+1} in parallel
    const int tn = t + 1;
    if (tn < TSTEPS) {
      for (int e = (bid - STEP_CVT_BLK0) * 256 + threadIdx.x;
           e < B_SEG * C_IN; e += (STEP_NBLKS - STEP_CVT_BLK0) * 256) {
        int b = e >> 7, c = e & (C_IN - 1);
        float v = (tn < lens[b]) ? x[(size_t)(offs[b] + tn) * C_IN + c] : 0.0f;
        xbf_next[e] = (__bf16)v;
      }
    }
    return;
  }

  // ---- GEMM + cell update (blocks 0..31, 8 waves each) ----
  const int wave = threadIdx.x >> 5;
  const int lane = threadIdx.x & 31;
  const int half = lane >> 4;
  const int l15  = lane & 15;
  const int idx  = bid * 8 + wave;              // 0..255
  const int bTile = idx >> 4;
  const int hTile = idx & 15;

  const int brow = bTile * 16 + l15;            // A row (batch)
  const int hcol = hTile * 16 + l15;            // B row = weight row (hidden col)

  const __bf16* xrow = xbf + (size_t)brow * C_IN;
  const __bf16* hrow = hbf + (size_t)brow * H_DIM;

  v8f acc[4];
#pragma unroll
  for (int g = 0; g < 4; ++g) acc[g] = (v8f){};

  // Phase 1: x_t @ W_ih^T (K = 128). A fragment reused across the 4 gates.
#pragma unroll
  for (int kk = 0; kk < C_IN / 32; ++kk) {
    v16bf a = frag_from_bf(xrow + kk * 32, half);
#pragma unroll
    for (int g = 0; g < 4; ++g) {
      const __bf16* wr = WihB + (size_t)(g * H_DIM + hcol) * C_IN;
      v16bf b = frag_from_bf(wr + kk * 32, half);
      acc[g] = __builtin_amdgcn_wmma_f32_16x16x32_bf16(false, a, false, b, (short)0,
                                                       acc[g], false, false);
    }
  }
  // Phase 2: h_{t-1} @ W_hh^T (K = 256).
#pragma unroll
  for (int kk = 0; kk < H_DIM / 32; ++kk) {
    v16bf a = frag_from_bf(hrow + kk * 32, half);
#pragma unroll
    for (int g = 0; g < 4; ++g) {
      const __bf16* wr = WhhB + (size_t)(g * H_DIM + hcol) * H_DIM;
      v16bf b = frag_from_bf(wr + kk * 32, half);
      acc[g] = __builtin_amdgcn_wmma_f32_16x16x32_bf16(false, a, false, b, (short)0,
                                                       acc[g], false, false);
    }
  }

  // Biases depend only on the gate column (fixed per lane).
  const float bi = bih[0 * H_DIM + hcol] + bhh[0 * H_DIM + hcol];
  const float bf = bih[1 * H_DIM + hcol] + bhh[1 * H_DIM + hcol];
  const float bg = bih[2 * H_DIM + hcol] + bhh[2 * H_DIM + hcol];
  const float bo = bih[3 * H_DIM + hcol] + bhh[3 * H_DIM + hcol];

#pragma unroll
  for (int r = 0; r < 8; ++r) {
    const int b = bTile * 16 + r + half * 8;    // D layout: VGPR r -> M=r / r+8
    const float gi = acc[0][r] + bi;
    const float gf = acc[1][r] + bf;
    const float gg = acc[2][r] + bg;
    const float go = acc[3][r] + bo;
    const float cold = c_in[(size_t)b * H_DIM + hcol];
    const float cnew = sigmoidf_(gf) * cold + sigmoidf_(gi) * tanhf(gg);
    const float hnew = sigmoidf_(go) * tanhf(cnew);
    c_out[(size_t)b * H_DIM + hcol]   = cnew;
    h_out[(size_t)b * H_DIM + hcol]   = hnew;
    hbf_out[(size_t)b * H_DIM + hcol] = (__bf16)hnew;   // bf16 shadow for next step
  }
}

// ---------------------------------------------------------------------------
extern "C" void kernel_launch(void* const* d_in, const int* in_sizes, int n_in,
                              void* d_out, int out_size, void* d_ws, size_t ws_size,
                              hipStream_t stream) {
  (void)in_sizes; (void)n_in; (void)out_size; (void)ws_size;

  const float* x    = (const float*)d_in[0];
  const int*   lens = (const int*)  d_in[1];
  const float* Wih  = (const float*)d_in[2];
  const float* Whh  = (const float*)d_in[3];
  const float* bih  = (const float*)d_in[4];
  const float* bhh  = (const float*)d_in[5];
  const float* h0W1 = (const float*)d_in[6];
  const float* h0W2 = (const float*)d_in[7];
  const float* c0W1 = (const float*)d_in[8];
  const float* c0W2 = (const float*)d_in[9];
  const float* fcW  = (const float*)d_in[10];
  const float* fcb  = (const float*)d_in[11];
  float* out = (float*)d_out;

  // Workspace carve-up (~2.7 MB total).
  char* ws = (char*)d_ws;
  int* offs = (int*)ws;          ws += 1024;
  float* sums = (float*)ws;      ws += (size_t)B_SEG * C_IN * 4;
  float* tmp  = (float*)ws;      ws += (size_t)B_SEG * C_IN * 4;
  float* hA   = (float*)ws;      ws += (size_t)B_SEG * H_DIM * 4;
  float* hB   = (float*)ws;      ws += (size_t)B_SEG * H_DIM * 4;
  float* cA   = (float*)ws;      ws += (size_t)B_SEG * H_DIM * 4;
  float* cB   = (float*)ws;      ws += (size_t)B_SEG * H_DIM * 4;
  float* fcin = (float*)ws;      ws += (size_t)B_SEG * H_DIM * 4;
  __bf16* WihB = (__bf16*)ws;    ws += (size_t)4 * H_DIM * C_IN * 2;
  __bf16* WhhB = (__bf16*)ws;    ws += (size_t)4 * H_DIM * H_DIM * 2;
  __bf16* xbfA = (__bf16*)ws;    ws += (size_t)B_SEG * C_IN * 2;
  __bf16* xbfB = (__bf16*)ws;    ws += (size_t)B_SEG * C_IN * 2;
  __bf16* hbfA = (__bf16*)ws;    ws += (size_t)B_SEG * H_DIM * 2;
  __bf16* hbfB = (__bf16*)ws;    /* end */

  prefix_kernel<<<1, 32, 0, stream>>>(lens, offs, B_SEG);
  segsum_kernel<<<B_SEG, C_IN, 0, stream>>>(x, offs, lens, sums);
  cvt_bf16_kernel<<<(4 * H_DIM * C_IN + 255) / 256, 256, 0, stream>>>(Wih, WihB, 4 * H_DIM * C_IN);
  cvt_bf16_kernel<<<(4 * H_DIM * H_DIM + 255) / 256, 256, 0, stream>>>(Whh, WhhB, 4 * H_DIM * H_DIM);

  // h0 / c0: relu(sums @ W1^T) @ W2^T
  {
    const int Nt1 = C_IN / 16,  tot1 = (B_SEG / 16) * Nt1;   // 128 tiles
    const int Nt2 = H_DIM / 16, tot2 = (B_SEG / 16) * Nt2;   // 256 tiles
    wmma_gemm_nt<<<(tot1 + 7) / 8, 256, 0, stream>>>(sums, C_IN, h0W1, C_IN, nullptr, tmp, C_IN, Nt1, tot1, C_IN, 1);
    wmma_gemm_nt<<<(tot2 + 7) / 8, 256, 0, stream>>>(tmp,  C_IN, h0W2, C_IN, nullptr, hA,  H_DIM, Nt2, tot2, C_IN, 0);
    wmma_gemm_nt<<<(tot1 + 7) / 8, 256, 0, stream>>>(sums, C_IN, c0W1, C_IN, nullptr, tmp, C_IN, Nt1, tot1, C_IN, 1);
    wmma_gemm_nt<<<(tot2 + 7) / 8, 256, 0, stream>>>(tmp,  C_IN, c0W2, C_IN, nullptr, cA,  H_DIM, Nt2, tot2, C_IN, 0);
  }

  // bf16 shadows of h0 and x_0.
  cvt_bf16_kernel<<<(B_SEG * H_DIM + 255) / 256, 256, 0, stream>>>(hA, hbfA, B_SEG * H_DIM);
  xcvt_kernel<<<(B_SEG * C_IN + 255) / 256, 256, 0, stream>>>(x, offs, lens, 0, xbfA);

  // Recurrent scan: only the first 256 steps feed the output.
  float*  hin = hA;   float*  hout = hB;
  float*  cin = cA;   float*  cout = cB;
  __bf16* hbfin = hbfA; __bf16* hbfout = hbfB;
  __bf16* xbfin = xbfA; __bf16* xbfnext = xbfB;
  for (int t = 0; t < TSTEPS; ++t) {
    lstm_step_kernel<<<STEP_NBLKS, 256, 0, stream>>>(
        xbfin, hbfin, hin, cin, WihB, WhhB, bih, bhh,
        hout, hbfout, cout, x, offs, lens, t, xbfnext, fcin);
    { float* s = hin; hin = hout; hout = s; }
    { float* s = cin; cin = cout; cout = s; }
    { __bf16* s = hbfin; hbfin = hbfout; hbfout = s; }
    { __bf16* s = xbfin; xbfin = xbfnext; xbfnext = s; }
  }
  // Last timestep's reduction (fused form only covers t-1 up to 254).
  reduce_h_kernel<<<1, H_DIM, 0, stream>>>(hin, lens, TSTEPS - 1, fcin);

  // Final FC: out[256,400] = fc_in @ fc_W^T + fc_b  (400 = 25 * 16, exact tiling)
  {
    const int Nt = NCLS / 16, tot = (B_SEG / 16) * Nt;       // 400 tiles
    wmma_gemm_nt<<<(tot + 7) / 8, 256, 0, stream>>>(fcin, H_DIM, fcW, H_DIM, fcb, out, NCLS, Nt, tot, H_DIM, 0);
  }
}
